// Attention_3917010174247
// MI455X (gfx1250) — compile-verified, matrix operand order checked by reading, not run on hardware
//
#include <hip/hip_runtime.h>

// ---------------------------------------------------------------------------
// Distance-softmax attention (N=4096, 16 bh-slices, d=64) for gfx1250.
//   1) qkv projection  -> Q (A-layout) / K,V (B-layout) f16 WMMA fragments
//   2) flash attention -> 8 waves/WG share one (b,h) slice; K/V fragment
//      blocks staged into double-buffered LDS by the Tensor Data Mover
//      (tensor_load_to_lds + s_wait_tensorcnt), online softmax over
//      sqrt(q2+k2-2qk), P re-layout D->A via per-wave LDS tile
//   3) output projection
// Workspace: qbuf 8MB + kbuf 8MB + vbuf 8MB + attn(f32) 16MB = 40MB (fits L2).
// ---------------------------------------------------------------------------

typedef __attribute__((ext_vector_type(16))) _Float16 v16h;
typedef __attribute__((ext_vector_type(8)))  float    v8f;
typedef __attribute__((ext_vector_type(4)))  unsigned int v4u;
typedef __attribute__((ext_vector_type(8)))  int      v8i;
typedef __attribute__((ext_vector_type(4)))  int      v4i;

#define NTOK   4096
#define CDIM   256
#define DINNER 512
#define NHEAD  8
#define DHEAD  64

#if defined(__has_builtin)
#  if __has_builtin(__builtin_amdgcn_tensor_load_to_lds)
#    define HAVE_TDM 1
#  else
#    define HAVE_TDM 0
#  endif
#else
#  define HAVE_TDM 0
#endif

union Frag16 { uint32_t u[8]; uint4 q[2]; v16h h; _Float16 e[16]; };
union AccF   { float f[8]; v8f v; };

static __device__ __forceinline__ uint32_t pack2(float a, float b) {
  union { _Float16 h[2]; uint32_t u; } r;
  r.h[0] = (_Float16)a; r.h[1] = (_Float16)b; return r.u;
}

#if HAVE_TDM
// 1-D TDM copy: nbytes (multiple of 8) from global to LDS.
// D# per CDNA5 ISA: group0 = {flags, lds_addr, gaddr,
// gaddr_hi|type=2}; group1 = {data_size=8B, tensor_dim0=tile_dim0=n8,
// tensor_dim1=1, dim0_stride=n8}. Groups 2/3 zero (<=2D tensor).
static __device__ __forceinline__ void tdm_load_1d(const void* gsrc,
                                                   uint32_t lds_off,
                                                   int nbytes) {
  const unsigned long long ga = (unsigned long long)(uintptr_t)gsrc;
  const unsigned n8 = (unsigned)nbytes >> 3;
  v4u g0;
  g0[0] = 1u;                                   // count=1 (valid user desc)
  g0[1] = lds_off;                              // lds_addr
  g0[2] = (unsigned)ga;                         // global_addr[31:0]
  g0[3] = (unsigned)(ga >> 32) | (2u << 30);    // global_addr[56:32] | type=2
  v8i g1;
  g1[0] = 0x00030000;                           // data_size=3 (8 bytes)
  g1[1] = (int)((n8 & 0xffffu) << 16);          // tensor_dim0[15:0]
  g1[2] = (int)((n8 >> 16) | (1u << 16));       // tensor_dim0[31:16] | dim1=1
  g1[3] = (int)(n8 << 16);                      // tile_dim0 = n8
  g1[4] = 0;
  g1[5] = (int)n8;                              // tensor_dim0_stride
  g1[6] = 0; g1[7] = 0;
  v4i z4 = {0, 0, 0, 0};
#if __clang_major__ >= 23
  v8i z8 = {0, 0, 0, 0, 0, 0, 0, 0};
  __builtin_amdgcn_tensor_load_to_lds(g0, g1, z4, z4, z8, 0);
#else
  __builtin_amdgcn_tensor_load_to_lds(g0, g1, z4, z4, 0);
#endif
}
#endif

// A-layout (16-bit, 16x32, MxK): lane = M + 16*((K>>3)&1),
//   vgpr = (K>>4)*4 + ((K>>1)&3), half = K&1.
// B-layout (16-bit, 32x16, KxN) mirrors with N across lanes.
// Fragment storage: 256 dwords; dword (lane,v) at lane*8+v  (2x b128 / lane).

// ------------------------------ kernel 1 -----------------------------------
__global__ __launch_bounds__(256) void qkv_kernel(
    const float* __restrict__ x,     // [B][256][4096]
    const float* __restrict__ wqkv,  // [1536][256]
    uint32_t* __restrict__ qbuf, uint32_t* __restrict__ kbuf,
    uint32_t* __restrict__ vbuf)
{
  const int wid  = (blockIdx.x * blockDim.x + threadIdx.x) >> 5;
  const int lane = threadIdx.x & 31;
  const int nt = wid & 255;
  const int ot = (wid >> 8) % 96;
  const int b  = wid / (96 * 256);
  const int m  = lane & 15, hi = lane >> 4;
  const int n0 = nt * 16;
  const int ocol = ot * 16 + m;            // output-channel column (N dim of D)
  const float* xb = x + (size_t)b * CDIM * NTOK;

  AccF acc;
#pragma unroll
  for (int i = 0; i < 8; ++i) acc.f[i] = 0.f;

#pragma unroll
  for (int cs = 0; cs < 8; ++cs) {         // K over C=256 in steps of 32
    Frag16 A, Bf;
#pragma unroll
    for (int v = 0; v < 8; ++v) {
      const int kb = (v >> 2) * 16 + hi * 8 + (v & 3) * 2 + cs * 32;
      float a0 = xb[(size_t)kb * NTOK + n0 + m];
      float a1 = xb[(size_t)(kb + 1) * NTOK + n0 + m];
      A.u[v] = pack2(a0, a1);
      const float2 wv = *(const float2*)(wqkv + (size_t)ocol * CDIM + kb);
      Bf.u[v] = pack2(wv.x, wv.y);
    }
    acc.v = __builtin_amdgcn_wmma_f32_16x16x32_f16(false, A.h, false, Bf.h,
                                                   (short)0, acc.v, false, false);
  }

  // Scatter D into pre-swizzled fragment buffers (q: A-layout, k/v: B-layout)
  const int sel  = ocol >> 9;              // 0:q 1:k 2:v  (tile never straddles)
  const int oo   = ocol & 511;
  const int head = oo >> 6, dd = oo & 63;
  const int bh   = b * NHEAD + head;
#pragma unroll
  for (int r = 0; r < 8; ++r) {
    const int n_g = n0 + r + 8 * hi;       // token index of this D row
    const _Float16 val = (_Float16)acc.f[r];
    if (sel == 0) {        // Q A-frag (bh, it, kg): M = n, K = dd
      const int k = dd & 31, kg = dd >> 5;
      const int lp = (n_g & 15) + 16 * ((k >> 3) & 1);
      const int vp = (k >> 4) * 4 + ((k >> 1) & 3);
      _Float16* p = (_Float16*)(qbuf + ((size_t)(bh * 256 + (n_g >> 4)) * 2 + kg) * 256
                                + lp * 8 + vp);
      p[k & 1] = val;
    } else if (sel == 1) { // K B-frag (bh, jt, ddg): K = dd, N = j
      const int k = dd & 31, ddg = dd >> 5;
      const int lp = (n_g & 15) + 16 * ((k >> 3) & 1);
      const int vp = (k >> 4) * 4 + ((k >> 1) & 3);
      _Float16* p = (_Float16*)(kbuf + ((size_t)(bh * 256 + (n_g >> 4)) * 2 + ddg) * 256
                                + lp * 8 + vp);
      p[k & 1] = val;
    } else {               // V B-frag (bh, jg, dt): K = j, N = dd
      const int k = n_g & 31, jg = n_g >> 5, dt = dd >> 4;
      const int lp = (dd & 15) + 16 * ((k >> 3) & 1);
      const int vp = (k >> 4) * 4 + ((k >> 1) & 3);
      _Float16* p = (_Float16*)(vbuf + ((size_t)(bh * 128 + jg) * 4 + dt) * 256
                                + lp * 8 + vp);
      p[k & 1] = val;
    }
  }
}

// ------------------------------ kernel 2 -----------------------------------
// 8 waves per block; all waves share one (b,h) slice and stage K/V through
// double-buffered LDS filled by the Tensor Data Mover.
__global__ __launch_bounds__(256) void attn_kernel(
    const uint32_t* __restrict__ qbuf, const uint32_t* __restrict__ kbuf,
    const uint32_t* __restrict__ vbuf, float* __restrict__ attn /*[B][N][512]*/)
{
  __shared__ __align__(16) float    Pld[8][16][34];   // per-wave P transpose
  __shared__ __align__(16) uint32_t kv[2][2048];      // [buf][K 4KB | V 4KB]

  const int lane  = threadIdx.x & 31;
  const int w     = threadIdx.x >> 5;
  const int slice = blockIdx.x >> 5;                  // 16 slices x 32 blocks
  const int it    = (blockIdx.x & 31) * 8 + w;        // this wave's 16-row strip
  const int hi = lane >> 4, lo = lane & 15;

  // Q A-fragments (resident for whole strip)
  Frag16 qA0, qA1;
  {
    const uint4* p0 = (const uint4*)(qbuf + ((size_t)(slice * 256 + it) * 2 + 0) * 256);
    const uint4* p1 = (const uint4*)(qbuf + ((size_t)(slice * 256 + it) * 2 + 1) * 256);
    qA0.q[0] = p0[lane * 2]; qA0.q[1] = p0[lane * 2 + 1];
    qA1.q[0] = p1[lane * 2]; qA1.q[1] = p1[lane * 2 + 1];
  }
  float q2p = 0.f;
#pragma unroll
  for (int i = 0; i < 16; ++i) {
    float a = (float)qA0.e[i], b2 = (float)qA1.e[i];
    q2p += a * a + b2 * b2;
  }
  q2p += __shfl_xor(q2p, 16, 32);
  float q2row[8];
#pragma unroll
  for (int r = 0; r < 8; ++r) q2row[r] = __shfl(q2p, r + 8 * hi, 32);

  float M[8], L[8];
#pragma unroll
  for (int r = 0; r < 8; ++r) { M[r] = 0.f; L[r] = 0.f; }   // sim >= 0 always
  AccF acc[4];
#pragma unroll
  for (int t = 0; t < 4; ++t)
#pragma unroll
    for (int r = 0; r < 8; ++r) acc[t].f[r] = 0.f;

  const uint32_t* kb_base = kbuf + (size_t)slice * 256 * 2 * 256;
  const uint32_t* vb_base = vbuf + (size_t)slice * 128 * 4 * 256;

#if HAVE_TDM
  if (w == 0) {            // TDM ignores EXEC; one op per wave. Preload jt=0.
    tdm_load_1d(kb_base, (uint32_t)(uintptr_t)&kv[0][0],    4096);
    tdm_load_1d(vb_base, (uint32_t)(uintptr_t)&kv[0][1024], 4096);
  }
#endif

  for (int jt = 0; jt < 128; ++jt) {       // 32 key columns per iteration
    const int pb = jt & 1;
#if HAVE_TDM
    if (w == 0) {
      if (jt + 1 < 128) {  // pipeline next block into the other buffer
        tdm_load_1d(kb_base + (size_t)(jt + 1) * 1024,
                    (uint32_t)(uintptr_t)&kv[1 - pb][0], 4096);
        tdm_load_1d(vb_base + (size_t)(jt + 1) * 1024,
                    (uint32_t)(uintptr_t)&kv[1 - pb][1024], 4096);
        __builtin_amdgcn_s_wait_tensorcnt(2);   // current buffer complete
      } else {
        __builtin_amdgcn_s_wait_tensorcnt(0);
      }
    }
    __syncthreads();                       // publish kv[pb] to all waves
#else
    for (int i = threadIdx.x; i < 1024; i += 256) {
      kv[pb][i]        = kb_base[(size_t)jt * 1024 + i];
      kv[pb][1024 + i] = vb_base[(size_t)jt * 1024 + i];
    }
    __syncthreads();
#endif

    // K fragments from LDS (ds_load_b128)
    Frag16 kf[2][2];                       // [j-half][dd-group]
#pragma unroll
    for (int jh = 0; jh < 2; ++jh)
#pragma unroll
      for (int g = 0; g < 2; ++g) {
        const uint4* p = (const uint4*)&kv[pb][(jh * 2 + g) * 256];
        kf[jh][g].q[0] = p[lane * 2]; kf[jh][g].q[1] = p[lane * 2 + 1];
      }

    AccF S0, S1;
#pragma unroll
    for (int r = 0; r < 8; ++r) { S0.f[r] = 0.f; S1.f[r] = 0.f; }
    S0.v = __builtin_amdgcn_wmma_f32_16x16x32_f16(false, qA0.h, false, kf[0][0].h, (short)0, S0.v, false, false);
    S0.v = __builtin_amdgcn_wmma_f32_16x16x32_f16(false, qA1.h, false, kf[0][1].h, (short)0, S0.v, false, false);
    S1.v = __builtin_amdgcn_wmma_f32_16x16x32_f16(false, qA0.h, false, kf[1][0].h, (short)0, S1.v, false, false);
    S1.v = __builtin_amdgcn_wmma_f32_16x16x32_f16(false, qA1.h, false, kf[1][1].h, (short)0, S1.v, false, false);

    // k2 per key column (lane holds column lo of each tile)
    float k20 = 0.f, k21 = 0.f;
#pragma unroll
    for (int i = 0; i < 16; ++i) {
      float a = (float)kf[0][0].e[i], b2 = (float)kf[0][1].e[i];
      float c = (float)kf[1][0].e[i], d = (float)kf[1][1].e[i];
      k20 += a * a + b2 * b2; k21 += c * c + d * d;
    }
    k20 += __shfl_xor(k20, 16, 32);
    k21 += __shfl_xor(k21, 16, 32);

    float sim0[8], sim1[8], rm[8];
#pragma unroll
    for (int r = 0; r < 8; ++r) {
      float t0 = q2row[r] + k20 - 2.f * S0.f[r];
      float t1 = q2row[r] + k21 - 2.f * S1.f[r];
      sim0[r] = __builtin_sqrtf(fmaxf(t0, 0.f));
      sim1[r] = __builtin_sqrtf(fmaxf(t1, 0.f));
      rm[r] = fmaxf(sim0[r], sim1[r]);
    }
#pragma unroll
    for (int off = 1; off < 16; off <<= 1)    // row max within 16-lane halves
#pragma unroll
      for (int r = 0; r < 8; ++r) rm[r] = fmaxf(rm[r], __shfl_xor(rm[r], off, 32));

    float P0[8], P1[8], ps[8], sc[8];
#pragma unroll
    for (int r = 0; r < 8; ++r) {
      float nm = fmaxf(M[r], rm[r]);
      sc[r] = __expf(M[r] - nm); M[r] = nm;
      P0[r] = __expf(sim0[r] - nm);
      P1[r] = __expf(sim1[r] - nm);
      ps[r] = P0[r] + P1[r];
    }
#pragma unroll
    for (int off = 1; off < 16; off <<= 1)    // row sum
#pragma unroll
      for (int r = 0; r < 8; ++r) ps[r] += __shfl_xor(ps[r], off, 32);
#pragma unroll
    for (int r = 0; r < 8; ++r) L[r] = L[r] * sc[r] + ps[r];
#pragma unroll
    for (int t = 0; t < 4; ++t)
#pragma unroll
      for (int r = 0; r < 8; ++r) acc[t].f[r] *= sc[r];

    // P: D-layout -> A-layout via per-wave LDS tile (same-wave DS is in-order)
#pragma unroll
    for (int r = 0; r < 8; ++r) {
      Pld[w][r + 8 * hi][lo]      = P0[r];
      Pld[w][r + 8 * hi][16 + lo] = P1[r];
    }
    __builtin_amdgcn_wave_barrier();
    Frag16 pa;
#pragma unroll
    for (int v = 0; v < 8; ++v) {
      const int kb2 = (v >> 2) * 16 + hi * 8 + (v & 3) * 2;
      const float2 pv = *(const float2*)&Pld[w][lo][kb2];
      pa.u[v] = pack2(pv.x, pv.y);
    }
    __builtin_amdgcn_wave_barrier();

    // V fragments from LDS; accumulate P x V
#pragma unroll
    for (int t = 0; t < 4; ++t) {
      Frag16 vf;
      const uint4* p = (const uint4*)&kv[pb][1024 + t * 256];
      vf.q[0] = p[lane * 2]; vf.q[1] = p[lane * 2 + 1];
      acc[t].v = __builtin_amdgcn_wmma_f32_16x16x32_f16(false, pa.h, false, vf.h,
                                                        (short)0, acc[t].v, false, false);
    }
    __syncthreads();       // buffer kv[pb] free before TDM refills it
  }

  // normalize + store [B][N][512] f32
  const int b = slice >> 3, head = slice & 7;
  float inv[8];
#pragma unroll
  for (int r = 0; r < 8; ++r) inv[r] = 1.f / L[r];
#pragma unroll
  for (int t = 0; t < 4; ++t)
#pragma unroll
    for (int r = 0; r < 8; ++r) {
      const int n_g = it * 16 + r + 8 * hi;
      attn[((size_t)b * NTOK + n_g) * DINNER + head * DHEAD + t * 16 + lo] =
          acc[t].f[r] * inv[r];
    }
}

// ------------------------------ kernel 3 -----------------------------------
__global__ __launch_bounds__(256) void proj_kernel(
    const float* __restrict__ attn,   // [B][4096][512]
    const float* __restrict__ wout,   // [256][512]
    float* __restrict__ out)          // [B][256][4096]
{
  const int wid  = (blockIdx.x * blockDim.x + threadIdx.x) >> 5;
  const int lane = threadIdx.x & 31;
  const int nt = wid & 255;
  const int ot = (wid >> 8) & 15;
  const int b  = wid >> 12;
  const int m = lane & 15, hi = lane >> 4;
  const int n0 = nt * 16;
  const int ocol = ot * 16 + m;
  const float* ab = attn + (size_t)b * NTOK * DINNER;

  AccF acc;
#pragma unroll
  for (int i = 0; i < 8; ++i) acc.f[i] = 0.f;

#pragma unroll 4
  for (int cs = 0; cs < 16; ++cs) {        // K over C2=512 in steps of 32
    Frag16 A, Bf;
#pragma unroll
    for (int v = 0; v < 8; ++v) {
      const int kb = (v >> 2) * 16 + hi * 8 + (v & 3) * 2 + cs * 32;
      const float2 av = *(const float2*)(ab + (size_t)(n0 + m) * DINNER + kb);
      A.u[v] = pack2(av.x, av.y);
      const float2 wv = *(const float2*)(wout + (size_t)ocol * DINNER + kb);
      Bf.u[v] = pack2(wv.x, wv.y);
    }
    acc.v = __builtin_amdgcn_wmma_f32_16x16x32_f16(false, A.h, false, Bf.h,
                                                   (short)0, acc.v, false, false);
  }
#pragma unroll
  for (int r = 0; r < 8; ++r) {
    const int n_g = n0 + r + 8 * hi;
    out[((size_t)b * CDIM + ocol) * NTOK + n_g] = acc.f[r];
  }
}

// ------------------------------ launcher -----------------------------------
extern "C" void kernel_launch(void* const* d_in, const int* in_sizes, int n_in,
                              void* d_out, int out_size, void* d_ws, size_t ws_size,
                              hipStream_t stream) {
  (void)in_sizes; (void)n_in; (void)out_size; (void)ws_size;
  const float* fmap = (const float*)d_in[0];   // [2,256,64,64]
  const float* wqkv = (const float*)d_in[1];   // [1536,256]
  const float* wout = (const float*)d_in[2];   // [256,512]
  float* out = (float*)d_out;                  // [2,256,64,64]

  uint32_t* qbuf = (uint32_t*)d_ws;                       // 8192 frags * 1KB
  uint32_t* kbuf = qbuf + (size_t)8192 * 256;             // 8 MB
  uint32_t* vbuf = kbuf + (size_t)8192 * 256;             // 8 MB
  float*    attn = (float*)(vbuf + (size_t)8192 * 256);   // 16 MB

  // 2*96*256 = 49152 wave-tiles, 8 waves / 256-thread block
  qkv_kernel<<<6144, 256, 0, stream>>>(fmap, wqkv, qbuf, kbuf, vbuf);
  // 16 slices * 32 blocks (8 waves each -> 256 strips/slice)
  attn_kernel<<<512, 256, 0, stream>>>(qbuf, kbuf, vbuf, attn);
  // 2*16*256 = 8192 wave-tiles
  proj_kernel<<<1024, 256, 0, stream>>>(attn, wout, out);
}
